// BatchSplitFF_18528488915208
// MI455X (gfx1250) — matrix operand order repaired and checked by reading, not compiled, not run on hardware
//
#include <hip/hip_runtime.h>
#include <hip/hip_bf16.h>

// ---------------------------------------------------------------------------
// BatchSplitFF for MI455X (gfx1250, wave32, WMMA bf16 w/ f32 accumulate)
//
// Shapes: B=4, SEQ=2048, DM=1024, t=32 tokens/group, 64 groups/ex -> NG=256
//         NSLOT = 32 experts * 4 sets = 128, F (expert hidden) = 32
// ---------------------------------------------------------------------------

#define DM      1024
#define SEQB    2048
#define NB      4
#define TGRP    32
#define GPEREX  64
#define NG      256      // NB * GPEREX
#define NSLOT   128
#define FF      32

typedef __attribute__((ext_vector_type(16))) __bf16 v16bf;
typedef __attribute__((ext_vector_type(8)))  __bf16 v8bf;
typedef __attribute__((ext_vector_type(4)))  __bf16 v4bf;
typedef __attribute__((ext_vector_type(8)))  float  v8f;

// ----- WMMA fragment helpers (per CDNA5 ISA 7.12.2 16-bit layouts) ----------
// A (16x32, row per lane%16): elems 0..7 = K base..base+7, 8..15 = K base+16..,
// base = 8 for lanes 16..31.
__device__ __forceinline__ v16bf frag_a_row(const __bf16* row, int kbase, int lane) {
  const int b0 = kbase + ((lane & 16) ? 8 : 0);
  v8bf lo = *(const v8bf*)(row + b0);
  v8bf hi = *(const v8bf*)(row + b0 + 16);
  v16bf r;
#pragma unroll
  for (int i = 0; i < 8; ++i) { r[i] = lo[i]; r[i + 8] = hi[i]; }
  return r;
}
// B (32x16, col per lane%16): same K order as A (elems 0..7 = K base..base+7,
// 8..15 = K base+16.., base = 8 for lanes 16..31).
__device__ __forceinline__ v16bf frag_b_col(const __bf16* row, int kbase, int lane) {
  const int b0 = kbase + ((lane & 16) ? 8 : 0);
  v8bf lo = *(const v8bf*)(row + b0);
  v8bf hi = *(const v8bf*)(row + b0 + 16);
  v16bf r;
#pragma unroll
  for (int i = 0; i < 8; ++i) { r[i] = lo[i]; r[i + 8] = hi[i]; }
  return r;
}
__device__ __forceinline__ v8f wmma_bf16(v16bf a, v16bf b, v8f c) {
  return __builtin_amdgcn_wmma_f32_16x16x32_bf16(false, a, false, b, (short)0, c,
                                                 false, false);
}

// ===========================================================================
// K1: per-group router. grid = 256 (one block per (b,g) group), block = 256.
// LDS: xg[32][1040] bf16 (66560B) | cl[128][80] bf16 (20480B) | lbuf[32][128] f32
// ===========================================================================
__global__ void k_logits_argmax(const float* __restrict__ x,
                                const float* __restrict__ ctrl,
                                __bf16* __restrict__ xbf,
                                int* __restrict__ sel) {
  extern __shared__ char smem[];
  __bf16* xg   = (__bf16*)smem;                 // 32 rows x 1040 (pad keeps 32B align)
  __bf16* cl   = (__bf16*)(smem + 66560);       // 128 rows x 80
  float*  lbuf = (float*)(smem + 87040);        // 32 x 128

  const int tid  = threadIdx.x;
  const int lane = tid & 31;
  const int w    = tid >> 5;                    // 8 waves
  const int G    = blockIdx.x;                  // group id 0..255
  const int rowbase = (G >> 6) * SEQB + (G & 63) * TGRP;  // token row of t=0

  // Stage group tokens to LDS as bf16 and spill a bf16 copy of x to global
  // (the whole grid covers all of x exactly once).
#pragma unroll 4
  for (int i = 0; i < 32; ++i) {
    const int idx = tid + i * 256;              // 8192 float4 total
    const int tt  = idx >> 8;
    const int c4  = idx & 255;
    const float4 v = ((const float4*)x)[(size_t)(rowbase + tt) * 256 + c4];
    v4bf bv;
    bv[0] = (__bf16)v.x; bv[1] = (__bf16)v.y;
    bv[2] = (__bf16)v.z; bv[3] = (__bf16)v.w;
    *(v4bf*)(xg + tt * 1040 + c4 * 4) = bv;
    *(v4bf*)(xbf + (size_t)(rowbase + tt) * DM + c4 * 4) = bv;
  }

  // Router GEMM: M=32 (2 tiles), N=128 (wave w -> cols w*16..+15), K=1024.
  v8f acc0 = {}; v8f acc1 = {};
  const __bf16* arow0 = xg + ((lane & 15)) * 1040;
  const __bf16* arow1 = xg + (16 + (lane & 15)) * 1040;
  const __bf16* brow  = cl + (w * 16 + (lane & 15)) * 80;

  for (int kc = 0; kc < DM; kc += 64) {
    __syncthreads();
    // load controller chunk (64 x 128 f32) transposed -> cl[slot][dk]
#pragma unroll 2
    for (int i = 0; i < 8; ++i) {
      const int idx = tid + i * 256;            // 2048 float4
      const int dk  = idx >> 5;
      const int s4  = (idx & 31) * 4;
      const float4 v = ((const float4*)ctrl)[(size_t)(kc + dk) * 32 + (idx & 31)];
      cl[(s4 + 0) * 80 + dk] = (__bf16)v.x;
      cl[(s4 + 1) * 80 + dk] = (__bf16)v.y;
      cl[(s4 + 2) * 80 + dk] = (__bf16)v.z;
      cl[(s4 + 3) * 80 + dk] = (__bf16)v.w;
    }
    __syncthreads();
#pragma unroll
    for (int kk = 0; kk < 64; kk += 32) {
      const v16bf b  = frag_b_col(brow, kk, lane);
      const v16bf a0 = frag_a_row(arow0, kc + kk, lane);
      const v16bf a1 = frag_a_row(arow1, kc + kk, lane);
      acc0 = wmma_bf16(a0, b, acc0);
      acc1 = wmma_bf16(a1, b, acc1);
    }
  }
  __syncthreads();

  // tie-break (linspace(0,1e-6,32) over token axis) + stage logits to LDS
  {
    const int nc  = w * 16 + (lane & 15);
    const int mhi = (lane & 16) ? 8 : 0;
#pragma unroll
    for (int r = 0; r < 8; ++r) {
      const int t0 = mhi + r;
      const int t1 = 16 + mhi + r;
      lbuf[t0 * NSLOT + nc] = acc0[r] + (float)t0 * (1e-6f / 31.0f);
      lbuf[t1 * NSLOT + nc] = acc1[r] + (float)t1 * (1e-6f / 31.0f);
    }
  }
  __syncthreads();

  // argmax over the 32 tokens for each of the 128 slots
  if (tid < NSLOT) {
    float best = lbuf[tid];
    int   bi   = 0;
    for (int tt = 1; tt < TGRP; ++tt) {
      const float v = lbuf[tt * NSLOT + tid];
      if (v > best) { best = v; bi = tt; }
    }
    sel[G * NSLOT + tid] = bi;
  }
}

// ===========================================================================
// K2: per-slot expert FFN. grid = (2 M-halves, 128 slots), block = 256.
//   stage1: H(128x32) = gather(Xbf)(128x1024) @ W1(1024x32), bias+ReLU
//   stage2: Yt(1024x128) = W2t(1024x32) @ Ht(32x128); atomic scatter-add to out
// LDS: wbuf 66560B (W1t [32][1040] then W2t [1024][32]) | ag 16384 | hl 8192 | orow 512
// ===========================================================================
__global__ void k_expert_ffn(const __bf16* __restrict__ xbf,
                             const int* __restrict__ sel,
                             const float* __restrict__ f1,
                             const float* __restrict__ bias,
                             const float* __restrict__ f2,
                             float* __restrict__ out) {
  extern __shared__ char smem[];
  __bf16* wbuf = (__bf16*)smem;                          // 66560 B (union W1t/W2t)
  __bf16* ag   = (__bf16*)(smem + 66560);                // 128 x 64 gathered A chunk
  __bf16* hl   = (__bf16*)(smem + 66560 + 16384);        // 128 x 32 H (bf16)
  int*    orow = (int*)(smem + 66560 + 16384 + 8192);    // 128 token-row indices

  const int tid   = threadIdx.x;
  const int lane  = tid & 31;
  const int w     = tid >> 5;
  const int slot  = blockIdx.y;            // 0..127  (e*4 + s)
  const int Gbase = blockIdx.x * 128;      // 0 or 128

  // selected-token row index per group in this half
  if (tid < 128) {
    const int G = Gbase + tid;
    const int s = sel[G * NSLOT + slot];
    orow[tid] = (G >> 6) * SEQB + (G & 63) * TGRP + s;
  }

  // W1 slice (d, f) at f1[(d*128+slot)*32 + f]  ->  wbuf[f][d] (rows 1040)
#pragma unroll 4
  for (int i = 0; i < 32; ++i) {
    const int idx = tid + i * 256;           // 8192 float4
    const int d   = idx >> 3;
    const int f4  = (idx & 7) * 4;
    const float4 v = *(const float4*)(f1 + ((size_t)(d * NSLOT + slot)) * FF + f4);
    wbuf[(f4 + 0) * 1040 + d] = (__bf16)v.x;
    wbuf[(f4 + 1) * 1040 + d] = (__bf16)v.y;
    wbuf[(f4 + 2) * 1040 + d] = (__bf16)v.z;
    wbuf[(f4 + 3) * 1040 + d] = (__bf16)v.w;
  }

  // ---- stage 1: M=128 (wave w -> M-tile w), N=32 (2 tiles), K=1024 ----
  v8f acc0 = {}; v8f acc1 = {};
  const __bf16* arow  = ag + (w * 16 + (lane & 15)) * 64;
  const __bf16* brow0 = wbuf + ((lane & 15)) * 1040;
  const __bf16* brow1 = wbuf + (16 + (lane & 15)) * 1040;

  for (int kc = 0; kc < DM; kc += 64) {
    __syncthreads();
    {   // gather 128 rows x 64 bf16 from the bf16 x copy (L2-resident)
      const int m = tid >> 1, half = tid & 1;
      const __bf16* srcp = xbf + (size_t)orow[m] * DM + kc + half * 32;
      const uint4* src = (const uint4*)srcp;
      uint4* dst = (uint4*)(ag + m * 64 + half * 32);
#pragma unroll
      for (int q = 0; q < 4; ++q) dst[q] = src[q];
      if (kc + 64 < DM) __builtin_prefetch(srcp + 64, 0, 1);
    }
    __syncthreads();
#pragma unroll
    for (int kk = 0; kk < 64; kk += 32) {
      const v16bf a  = frag_a_row(arow, kk, lane);
      const v16bf b0 = frag_b_col(brow0, kc + kk, lane);
      const v16bf b1 = frag_b_col(brow1, kc + kk, lane);
      acc0 = wmma_bf16(a, b0, acc0);
      acc1 = wmma_bf16(a, b1, acc1);
    }
  }

  // bias + ReLU -> hl[m][f] (bf16)
  {
    const float bv0 = bias[slot * FF + (lane & 15)];
    const float bv1 = bias[slot * FF + 16 + (lane & 15)];
    const int mbase = w * 16 + ((lane & 16) ? 8 : 0);
    const int fc    = (lane & 15);
#pragma unroll
    for (int r = 0; r < 8; ++r) {
      float h0 = acc0[r] + bv0; h0 = h0 > 0.f ? h0 : 0.f;
      float h1 = acc1[r] + bv1; h1 = h1 > 0.f ? h1 : 0.f;
      hl[(mbase + r) * FF + fc]      = (__bf16)h0;
      hl[(mbase + r) * FF + 16 + fc] = (__bf16)h1;
    }
  }
  __syncthreads();

  // W2 slice (f, d) at f2[(slot*32+f)*1024 + d]  ->  wbuf[d][f] (rows 32)
#pragma unroll 4
  for (int i = 0; i < 32; ++i) {
    const int idx = tid + i * 256;            // 8192 float4, d-contiguous reads
    const int f   = idx >> 8;
    const int d4  = (idx & 255) * 4;
    const float4 v = *(const float4*)(f2 + ((size_t)slot * FF + f) * DM + d4);
    wbuf[(d4 + 0) * FF + f] = (__bf16)v.x;
    wbuf[(d4 + 1) * FF + f] = (__bf16)v.y;
    wbuf[(d4 + 2) * FF + f] = (__bf16)v.z;
    wbuf[(d4 + 3) * FF + f] = (__bf16)v.w;
  }
  __syncthreads();

  // ---- stage 2: Yt = W2t(1024x32) @ Ht(32x128); wave w -> N-tile w ----
  const v16bf bfrag = frag_b_col(hl + (w * 16 + (lane & 15)) * FF, 0, lane);
  const int   n     = w * 16 + (lane & 15);
  float* obase      = out + (size_t)orow[n] * DM;
  const int   dhi   = (lane & 16) ? 8 : 0;

  for (int mt = 0; mt < 64; ++mt) {
    const v16bf a = frag_a_row(wbuf + (mt * 16 + (lane & 15)) * FF, 0, lane);
    v8f c = {};
    c = wmma_bf16(a, bfrag, c);
    const int dbase = mt * 16 + dhi;
#pragma unroll
    for (int r = 0; r < 8; ++r) atomicAdd(obase + dbase + r, c[r]);
  }
}

// ===========================================================================
// Host launcher
// ===========================================================================
extern "C" void kernel_launch(void* const* d_in, const int* in_sizes, int n_in,
                              void* d_out, int out_size, void* d_ws, size_t ws_size,
                              hipStream_t stream) {
  (void)in_sizes; (void)n_in; (void)ws_size;
  const float* x    = (const float*)d_in[0];   // (4, 2048, 1024)
  const float* ctrl = (const float*)d_in[1];   // (1024, 32, 4)
  const float* f1   = (const float*)d_in[2];   // (1024, 32, 4, 32)
  const float* bias = (const float*)d_in[3];   // (32, 4, 32)
  const float* f2   = (const float*)d_in[4];   // (32, 4, 32, 1024)
  float* out = (float*)d_out;                  // (4, 2048, 1024)

  // workspace: bf16 copy of x (16 MB) + sel indices (128 KB)
  __bf16* xbf = (__bf16*)d_ws;
  int* sel = (int*)((char*)d_ws + (size_t)NB * SEQB * DM * sizeof(__bf16));

  const int lds1 = 66560 + 20480 + 16384;          // 103424 B
  const int lds2 = 66560 + 16384 + 8192 + 512;     //  91648 B
  static bool attr_done = false;                   // idempotent attr set (host-side)
  if (!attr_done) {
    hipFuncSetAttribute((const void*)k_logits_argmax,
                        hipFuncAttributeMaxDynamicSharedMemorySize, lds1);
    hipFuncSetAttribute((const void*)k_expert_ffn,
                        hipFuncAttributeMaxDynamicSharedMemorySize, lds2);
    attr_done = true;
  }

  // zero the output (scatter-add target); memset nodes are graph-capturable
  hipMemsetAsync(d_out, 0, (size_t)out_size * sizeof(float), stream);

  k_logits_argmax<<<NG, 256, lds1, stream>>>(x, ctrl, xbf, sel);
  k_expert_ffn<<<dim3(2, NSLOT), 256, lds2, stream>>>(xbf, sel, f1, bias, f2, out);
}
